// FaceProj_67783173865692
// MI455X (gfx1250) — matrix operand order repaired and checked by reading, not compiled, not run on hardware
//
#include <hip/hip_runtime.h>

typedef __attribute__((ext_vector_type(16))) __bf16 v16bf;
typedef __attribute__((ext_vector_type(8)))  __bf16 v8bf;
typedef __attribute__((ext_vector_type(4)))  __bf16 v4bf;
typedef __attribute__((ext_vector_type(8)))  float  v8f;

union BFrag { v16bf v; v8bf h[2]; };

#define T_TRIP   131072
#define FEAT     256
#define K0       1024      // padded input dim (970 -> 1024)
#define HID      512
#define ODIM     256
#define MT       64        // triplets per block (4 WMMA M-tiles, 2 M-pairs)
#define NTHREADS 256       // 8 waves of 32

// workspace layout in bf16 elements
#define W0B_OFF  0
#define W1B_OFF  (512 * 1024)
#define W2B_OFF  (W1B_OFF + 512 * 512)
#define W3B_OFF  (W2B_OFF + 512 * 512)
#define WOB_OFF  (W3B_OFF + 512 * 512)
#define WTOT     (WOB_OFF + 256 * 512)   // 1,441,792 elems (~2.9 MB)

// ---------------------------------------------------------------------------
// Convert all weights to bf16, pad W0 K-dim 970 -> 1024 with zeros.
// ---------------------------------------------------------------------------
__global__ __launch_bounds__(NTHREADS) void prep_weights_kernel(
    const float* __restrict__ W0, const float* __restrict__ W1,
    const float* __restrict__ W2, const float* __restrict__ W3,
    const float* __restrict__ Wout, __bf16* __restrict__ dst)
{
    int i = blockIdx.x * NTHREADS + threadIdx.x;
    if (i >= WTOT) return;
    float v;
    if (i < W1B_OFF) {                       // W0b [512][1024]
        int n = i >> 10, k = i & 1023;
        v = (k < 970) ? W0[n * 970 + k] : 0.0f;
    } else if (i < WOB_OFF) {                // W1b/W2b/W3b [512][512]
        int j = i - W1B_OFF;
        int L = j >> 18;                     // 0,1,2
        int r = j & (512 * 512 - 1);
        const float* Wp = (L == 0) ? W1 : (L == 1) ? W2 : W3;
        v = Wp[r];
    } else {                                 // Woutb [256][512]
        v = Wout[i - WOB_OFF];
    }
    dst[i] = (__bf16)v;
}

// ---------------------------------------------------------------------------
// 32x32 output patch per wave via 2x2 WMMA register blocking, K steps of 32.
//   in_lds : [MT][K] bf16 row-major
//   Wmat   : [Nout][K] bf16 row-major (global; B[k,n] = W[n,k])
// A layout (16-bit, 16x32): lanes 0-15 -> M=lane, v0-7 = K k..k+15 (contiguous)
//                           lanes 16-31 -> M=lane-16, v0-7 = K k+16..k+31
// B layout (32x16): lanes 0-15 -> N=lane, v0-7 = K k..k+15 (contiguous)
//                   lanes 16-31 -> N=lane-16, v0-7 = K k+16..k+31
// ---------------------------------------------------------------------------
__device__ __forceinline__ void store_tile(
    const v8f& acc, int mrow, int ncol, int rbias, int Nout,
    __bf16* out_lds, float* out_global, int tbase)
{
    if (out_global) {
#pragma unroll
        for (int r = 0; r < 8; ++r)
            out_global[(size_t)(tbase + mrow + r + rbias) * ODIM + ncol] = acc[r];
    } else {
#pragma unroll
        for (int r = 0; r < 8; ++r) {
            float v = acc[r];
            v = v / (1.0f + __expf(-v));              // silu
            out_lds[(mrow + r + rbias) * Nout + ncol] = (__bf16)v;
        }
    }
}

__device__ __forceinline__ void mlp_gemm(
    const __bf16* __restrict__ Wmat, const __bf16* in_lds, int K, int Nout,
    __bf16* out_lds,        // non-null: apply silu, store bf16 [MT][Nout]
    float* out_global,      // non-null: final layer, store f32 [T][ODIM]
    int tbase)
{
    const int tid  = threadIdx.x;
    const int lane = tid & 31;
    const int wid  = tid >> 5;
    const bool lo  = lane < 16;
    const int l15  = lane & 15;

    const int mpairs = MT / 32;              // 2
    const int npairs = Nout / 32;
    const int patches = mpairs * npairs;     // 32 (hidden) / 16 (out)

    const int ka0 = lo ? 0 : 16;
    const int ka1 = lo ? 8 : 24;
    const int kb0 = lo ? 0 : 16;
    const int rbias = lo ? 0 : 8;

    for (int patch = wid; patch < patches; patch += 8) {
        const int mb = (patch & (mpairs - 1)) * 32;
        const int nb = (patch / mpairs) * 32;

        const __bf16* arow0 = in_lds + (mb + l15) * K;
        const __bf16* arow1 = arow0 + 16 * K;
        const __bf16* brow0 = Wmat + (nb + l15) * K + kb0;
        const __bf16* brow1 = brow0 + 16 * K;

        v8f a00 = {}, a01 = {}, a10 = {}, a11 = {};
        // Bounded unroll: keeps A-fragment live ranges short so nothing spills
        // to scratch, while exposing 16 independent WMMAs per unrolled group.
#pragma unroll 4
        for (int k = 0; k < K; k += 32) {
            BFrag A0, A1, B0, B1;
            // global (L2) B loads first: longest latency
            B0.v    = *(const v16bf*)(brow0 + k);
            B1.v    = *(const v16bf*)(brow1 + k);
            // LDS A loads
            A0.h[0] = *(const v8bf*)(arow0 + k + ka0);
            A0.h[1] = *(const v8bf*)(arow0 + k + ka1);
            A1.h[0] = *(const v8bf*)(arow1 + k + ka0);
            A1.h[1] = *(const v8bf*)(arow1 + k + ka1);
            a00 = __builtin_amdgcn_wmma_f32_16x16x32_bf16(
                false, A0.v, false, B0.v, (short)0, a00, false, false);
            a01 = __builtin_amdgcn_wmma_f32_16x16x32_bf16(
                false, A0.v, false, B1.v, (short)0, a01, false, false);
            a10 = __builtin_amdgcn_wmma_f32_16x16x32_bf16(
                false, A1.v, false, B0.v, (short)0, a10, false, false);
            a11 = __builtin_amdgcn_wmma_f32_16x16x32_bf16(
                false, A1.v, false, B1.v, (short)0, a11, false, false);
        }

        store_tile(a00, mb,      nb      + l15, rbias, Nout, out_lds, out_global, tbase);
        store_tile(a01, mb,      nb + 16 + l15, rbias, Nout, out_lds, out_global, tbase);
        store_tile(a10, mb + 16, nb      + l15, rbias, Nout, out_lds, out_global, tbase);
        store_tile(a11, mb + 16, nb + 16 + l15, rbias, Nout, out_lds, out_global, tbase);
    }
}

// ---------------------------------------------------------------------------
// Fused: gather + RBF embed + 5-layer MLP for MT triplets per block.
// ---------------------------------------------------------------------------
__global__ __launch_bounds__(NTHREADS) void mlp_fused_kernel(
    const float* __restrict__ h, const int* __restrict__ trip,
    const float* __restrict__ nij, const float* __restrict__ nik,
    const float* __restrict__ cosv, const float* __restrict__ sinv,
    const __bf16* __restrict__ wb, float* __restrict__ out)
{
    __shared__ __bf16 bufA[MT * K0];   // 128 KB: layer-0 input, then 512-wide ping
    __shared__ __bf16 bufB[MT * HID];  //  64 KB: pong

    const int tbase = blockIdx.x * MT;
    const int tid = threadIdx.x;

    // ---- stage 1: build bf16 input tile [MT][K0] in LDS, 4 cols per chunk ----
    // row layout: [0,256) h[src] | [256,512) h[dst_i] | [512,768) h[dst_j]
    //             [768,868) rbf(norm_ij) | [868,968) rbf(norm_ik)
    //             [968]=cos [969]=sin | [970,1024)=0
    // All region boundaries are multiples of 4, so 4-col chunks never straddle.
#pragma unroll 2
    for (int i = 0; i < (MT * K0 / 4) / NTHREADS; ++i) {
        int c    = tid + i * NTHREADS;       // chunk id
        int row  = c >> 8;                   // 256 chunks per row
        int col4 = (c & 255) * 4;
        int t    = tbase + row;
        v4bf val;
        if (col4 < 768) {
            int which = col4 >> 8;                 // 0:src 1:dst_i 2:dst_j
            int f     = col4 & 255;
            int node  = trip[which * T_TRIP + t];
            const float4 hv = *(const float4*)(h + (size_t)node * FEAT + f);
            val[0] = (__bf16)hv.x; val[1] = (__bf16)hv.y;
            val[2] = (__bf16)hv.z; val[3] = (__bf16)hv.w;
        } else if (col4 < 968) {
            const bool first = col4 < 868;
            const float d  = first ? nij[t] : nik[t];
            const float m0 = 0.1f * (float)(col4 - (first ? 768 : 868));
#pragma unroll
            for (int j = 0; j < 4; ++j) {
                float diff = m0 + 0.1f * (float)j - d;
                val[j] = (__bf16)__expf(-10.0f * diff * diff);
            }
        } else if (col4 == 968) {
            val[0] = (__bf16)cosv[t]; val[1] = (__bf16)sinv[t];
            val[2] = (__bf16)0.0f;    val[3] = (__bf16)0.0f;
        } else {
            val[0] = val[1] = val[2] = val[3] = (__bf16)0.0f;
        }
        *(v4bf*)(&bufA[(row << 10) + col4]) = val;
    }
    __syncthreads();

    // ---- stage 2: MLP ----
    mlp_gemm(wb + W0B_OFF, bufA, K0,  HID, bufB, nullptr, tbase);  // 1024->512 silu
    __syncthreads();
    mlp_gemm(wb + W1B_OFF, bufB, HID, HID, bufA, nullptr, tbase);  // 512->512 silu
    __syncthreads();
    mlp_gemm(wb + W2B_OFF, bufA, HID, HID, bufB, nullptr, tbase);
    __syncthreads();
    mlp_gemm(wb + W3B_OFF, bufB, HID, HID, bufA, nullptr, tbase);
    __syncthreads();
    mlp_gemm(wb + WOB_OFF, bufA, HID, ODIM, nullptr, out, tbase);  // 512->256, no act
}

extern "C" void kernel_launch(void* const* d_in, const int* in_sizes, int n_in,
                              void* d_out, int out_size, void* d_ws, size_t ws_size,
                              hipStream_t stream) {
    const float* h    = (const float*)d_in[0];
    const int*   trip = (const int*)d_in[1];
    const float* nij  = (const float*)d_in[2];
    const float* nik  = (const float*)d_in[3];
    const float* cosv = (const float*)d_in[4];
    const float* sinv = (const float*)d_in[5];
    // d_in[6] = mu (recomputed on device: mu[k] = 0.1*k)
    const float* W0   = (const float*)d_in[7];
    const float* W1   = (const float*)d_in[8];
    const float* W2   = (const float*)d_in[9];
    const float* W3   = (const float*)d_in[10];
    const float* Wout = (const float*)d_in[11];

    __bf16* wb = (__bf16*)d_ws;

    int prep_blocks = (WTOT + NTHREADS - 1) / NTHREADS;
    prep_weights_kernel<<<prep_blocks, NTHREADS, 0, stream>>>(W0, W1, W2, W3, Wout, wb);

    int mlp_blocks = T_TRIP / MT;   // 2048
    mlp_fused_kernel<<<mlp_blocks, NTHREADS, 0, stream>>>(
        h, trip, nij, nik, cosv, sinv, wb, (float*)d_out);
}
